// GroupLinear_87067577025281
// MI455X (gfx1250) — matrix operand order repaired and checked by reading, not compiled, not run on hardware
//
#include <hip/hip_runtime.h>

typedef __attribute__((ext_vector_type(16))) __bf16 bf16x16;
typedef __attribute__((ext_vector_type(8)))  float  f32x8;

#define IN_CH   4096
#define OUT_CH  4096
#define GROUPS  64
#define GSIZE   64            // IN_SCALE == OUT_SCALE == 64
#define NTOK    8192
#define M_WAVE  32            // two 16-row WMMA sub-tiles per wave
#define WAVES   8
#define M_BLOCK (M_WAVE * WAVES)   // 256 tokens per block

#define W_ELEMS (GROUPS * GSIZE * GSIZE)   // 262144 diagonal weight elements

// ---------------------------------------------------------------------------
// Prep: split the 64x64 diagonal blocks of W into bf16 hi/lo, stored in the
// exact per-lane WMMA B-operand order:
//   flat = (((g*4 + nt)*2 + kc)*32 + lane)*16 + slot
//   o = nt*16 + (lane&15),  K = kc*32 + (lane>>4)*16 + slot
// ---------------------------------------------------------------------------
__global__ __launch_bounds__(256)
void prep_w_split(const float* __restrict__ w,
                  __bf16* __restrict__ whi, __bf16* __restrict__ wlo)
{
    const int idx = blockIdx.x * 256 + threadIdx.x;     // 0 .. W_ELEMS-1
    const int s    = idx & 15;
    const int lane = (idx >> 4) & 31;
    const int kc   = (idx >> 9) & 1;
    const int nt   = (idx >> 10) & 3;
    const int g    = idx >> 12;

    const int o = nt * 16 + (lane & 15);
    const int k = kc * 32 + (lane >> 4) * 16 + s;

    const float  f = w[(size_t)(g * GSIZE + o) * IN_CH + (size_t)g * GSIZE + k];
    const __bf16 h = (__bf16)f;                 // RNE fp32->bf16
    whi[idx] = h;
    wlo[idx] = (__bf16)(f - (float)h);
}

// ---------------------------------------------------------------------------
// Main GEMM: one wave computes a 32-token x 64-output strip of one group.
// Two 16-row A sub-tiles share every B-operand load (halves L2 B traffic).
// PRECONV=true  -> B operands loaded directly as bf16 from d_ws (no converts)
// PRECONV=false -> self-contained fallback (converts W in-kernel)
// ---------------------------------------------------------------------------
template<bool PRECONV>
__global__ __launch_bounds__(256)
void group_linear_wmma_bf16x3(const float*  __restrict__ x,
                              const float*  __restrict__ w,
                              const __bf16* __restrict__ whi,
                              const __bf16* __restrict__ wlo,
                              float* __restrict__ y)
{
    const int lane = threadIdx.x & 31;
    const int wave = threadIdx.x >> 5;
    const int lm   = lane & 15;     // matrix row (A) / col (B,C)
    const int lh   = lane >> 4;     // lane-half select
    const int g    = blockIdx.y;
    const int m0   = blockIdx.x * M_BLOCK + wave * M_WAVE;

    // ---- A operands: x[m0 + mi*16 + lm, g*64 + k], split bf16 hi/lo ---------
    // A k order used here (same as B): lane half 0 -> K {kc*32..kc*32+15},
    // lane half 1 -> same +16.  One contiguous 16-float run per lane.
    bf16x16 a_hi[2][2], a_lo[2][2];          // [mi][kc]
#pragma unroll
    for (int mi = 0; mi < 2; ++mi) {
        const float* xrow = x + (size_t)(m0 + mi * 16 + lm) * IN_CH
                              + (size_t)g * GSIZE;
#pragma unroll
        for (int kc = 0; kc < 2; ++kc) {
            const int kb = kc * 32 + lh * 16;
            float v[16];
            *(float4*)(&v[0])  = *(const float4*)(xrow + kb + 0);
            *(float4*)(&v[4])  = *(const float4*)(xrow + kb + 4);
            *(float4*)(&v[8])  = *(const float4*)(xrow + kb + 8);
            *(float4*)(&v[12]) = *(const float4*)(xrow + kb + 12);
#pragma unroll
            for (int s = 0; s < 16; ++s) {
                float  f = v[s];
                __bf16 h = (__bf16)f;
                a_hi[mi][kc][s] = h;
                a_lo[mi][kc][s] = (__bf16)(f - (float)h);
            }
        }
    }

    // ---- 4 output 16-col tiles; B loaded once, used by both sub-tiles -------
#pragma unroll
    for (int nt = 0; nt < 4; ++nt) {
        bf16x16 b_hi[2], b_lo[2];
        if (PRECONV) {
            // operand-ready bf16: one 32 B load per operand per kc
#pragma unroll
            for (int kc = 0; kc < 2; ++kc) {
                const size_t ob = (size_t)((((g * 4 + nt) * 2 + kc) * 32 + lane) * 16);
                b_hi[kc] = *(const bf16x16*)(whi + ob);
                b_lo[kc] = *(const bf16x16*)(wlo + ob);
            }
        } else {
            // fallback: convert f32 weights in-kernel
            const float* wrow = w + (size_t)(g * GSIZE + nt * 16 + lm) * IN_CH
                                  + (size_t)g * GSIZE;
#pragma unroll
            for (int kc = 0; kc < 2; ++kc) {
                const int kb = kc * 32 + lh * 16;
                float v[16];
                *(float4*)(&v[0])  = *(const float4*)(wrow + kb + 0);
                *(float4*)(&v[4])  = *(const float4*)(wrow + kb + 4);
                *(float4*)(&v[8])  = *(const float4*)(wrow + kb + 8);
                *(float4*)(&v[12]) = *(const float4*)(wrow + kb + 12);
#pragma unroll
                for (int s = 0; s < 16; ++s) {
                    float  f = v[s];
                    __bf16 h = (__bf16)f;
                    b_hi[kc][s] = h;
                    b_lo[kc][s] = (__bf16)(f - (float)h);
                }
            }
        }

#pragma unroll
        for (int mi = 0; mi < 2; ++mi) {
            // y = xhi*whi + xlo*whi + xhi*wlo  (bf16x3 ~ fp32 accuracy)
            f32x8 acc = {};
            acc = __builtin_amdgcn_wmma_f32_16x16x32_bf16(false, a_hi[mi][0], false, b_hi[0], (short)0, acc, false, false);
            acc = __builtin_amdgcn_wmma_f32_16x16x32_bf16(false, a_hi[mi][1], false, b_hi[1], (short)0, acc, false, false);
            acc = __builtin_amdgcn_wmma_f32_16x16x32_bf16(false, a_lo[mi][0], false, b_hi[0], (short)0, acc, false, false);
            acc = __builtin_amdgcn_wmma_f32_16x16x32_bf16(false, a_lo[mi][1], false, b_hi[1], (short)0, acc, false, false);
            acc = __builtin_amdgcn_wmma_f32_16x16x32_bf16(false, a_hi[mi][0], false, b_lo[0], (short)0, acc, false, false);
            acc = __builtin_amdgcn_wmma_f32_16x16x32_bf16(false, a_hi[mi][1], false, b_lo[1], (short)0, acc, false, false);

            // C/D 16x16 f32 layout: VGPR r -> row m0 + mi*16 + r + lh*8,
            //                       col nt*16 + lm
            float* ybase = y + (size_t)(m0 + mi * 16 + lh * 8) * OUT_CH
                             + (size_t)g * GSIZE + nt * 16 + lm;
#pragma unroll
            for (int r = 0; r < 8; ++r)
                ybase[(size_t)r * OUT_CH] = acc[r];
        }
    }
}

extern "C" void kernel_launch(void* const* d_in, const int* in_sizes, int n_in,
                              void* d_out, int out_size, void* d_ws, size_t ws_size,
                              hipStream_t stream)
{
    const float* x = (const float*)d_in[0];
    const float* w = (const float*)d_in[1];
    float*       y = (float*)d_out;

    dim3 grid(NTOK / M_BLOCK, GROUPS);   // 32 x 64 blocks
    dim3 block(256);                      // 8 waves (wave32)

    const size_t need = (size_t)W_ELEMS * 2 * sizeof(__bf16);   // hi + lo = 1 MB
    if (ws_size >= need) {
        __bf16* whi = (__bf16*)d_ws;
        __bf16* wlo = whi + W_ELEMS;
        prep_w_split<<<dim3(W_ELEMS / 256), block, 0, stream>>>(w, whi, wlo);
        group_linear_wmma_bf16x3<true><<<grid, block, 0, stream>>>(x, w, whi, wlo, y);
    } else {
        group_linear_wmma_bf16x3<false><<<grid, block, 0, stream>>>(x, w, nullptr, nullptr, y);
    }
}
